// BertSelfAttention_2044404433178
// MI455X (gfx1250) — matrix operand order repaired and checked by reading, not compiled, not run on hardware
//
#include <hip/hip_runtime.h>

// ---------------------------------------------------------------------------
// BERT self-attention with relative_key_query bias, CDNA5 (gfx1250, wave32).
// All matrix math on v_wmma_f32_16x16x32_f16 (f16 operands, f32 accumulate).
// Bias einsums rewritten as Toeplitz GEMMs against dist_emb (Qd / Kd tiles).
// ---------------------------------------------------------------------------

typedef __attribute__((ext_vector_type(16))) _Float16 v16h;
typedef __attribute__((ext_vector_type(8)))  float    v8f;

union V16H { v16h v; float4 q[2]; };

static __device__ __forceinline__ v8f wmma_f16(v16h a, v16h b, v8f c) {
  return __builtin_amdgcn_wmma_f32_16x16x32_f16(false, a, false, b, (short)0, c,
                                                false, false);
}

#define S_LEN 1024
#define HDIM  1024
#define NH    16
#define HS    64

// workspace layout, offsets in _Float16 elements (~40.1 MB total)
#define XH_OFF   0u
#define WQ_OFF   (XH_OFF + 4096u * 1024u)
#define WK_OFF   (WQ_OFF + 1024u * 1024u)
#define WV_OFF   (WK_OFF + 1024u * 1024u)
#define E_OFF    (WV_OFF + 1024u * 1024u)     // 2047 rows + 1 zero pad row
#define QH_OFF   (E_OFF + 2048u * 64u)
#define KH_OFF   (QH_OFF + 4096u * 1024u)
#define VT_OFF   (KH_OFF + 4096u * 1024u)

// ---------------------------------------------------------------------------
// Stage 0: f32 -> f16 conversion (pads tail with zeros).
// ---------------------------------------------------------------------------
__global__ void cvt_f32_f16(const float* __restrict__ s,
                            _Float16* __restrict__ d, int n, int npad) {
  int i = blockIdx.x * 256 + threadIdx.x;
  if (i < npad) d[i] = (i < n) ? (_Float16)s[i] : (_Float16)0.0f;
}

// ---------------------------------------------------------------------------
// Stage 1: QKV projection  out[m,n] = sum_k X[m,k] * W[n,k] + b[n]
// grid (64 m-blocks, 16 heads, 3 matrices), 128 threads (4 waves).
// Wave w owns rows m0+16w .. +15, all 64 head columns.
// Q,K stored head-major [bh][S][64] f16; V stored transposed [bh][64][S] f16.
// ---------------------------------------------------------------------------
__global__ void __launch_bounds__(128)
qkv_gemm(const _Float16* __restrict__ X,  const _Float16* __restrict__ Wq,
         const _Float16* __restrict__ Wk, const _Float16* __restrict__ Wv,
         const float* __restrict__ bq, const float* __restrict__ bk,
         const float* __restrict__ bv,
         _Float16* __restrict__ Qh, _Float16* __restrict__ Kh,
         _Float16* __restrict__ Vt) {
  const int wave = threadIdx.x >> 5, lane = threadIdx.x & 31;
  const int ln = lane & 15, hi = lane >> 4;
  const int m0 = blockIdx.x * 64 + wave * 16;
  const int h  = blockIdx.y;
  const int nb = h * 64;
  const int z  = blockIdx.z;
  const _Float16* W = (z == 0) ? Wq : ((z == 1) ? Wk : Wv);
  const float* bias = (z == 0) ? bq : ((z == 1) ? bk : bv);

  v8f acc[4] = {};
  const _Float16* xrow = X + (size_t)(m0 + ln) * HDIM;
  for (int kc = 0; kc < HDIM; kc += 32) {
    V16H a;  // A operand: lane ln holds row m0+ln, K kc+hi*16+{0..15}
    a.q[0] = *(const float4*)(xrow + kc + hi * 16);
    a.q[1] = *(const float4*)(xrow + kc + hi * 16 + 8);
#pragma unroll
    for (int t = 0; t < 4; ++t) {
      // B[k,n] = W[n,k]: lane holds 16 contiguous k of W row n
      const _Float16* wrow = W + (size_t)(nb + t * 16 + ln) * HDIM + kc + hi * 16;
      V16H bm;
      bm.q[0] = *(const float4*)(wrow);
      bm.q[1] = *(const float4*)(wrow + 8);
      acc[t] = wmma_f16(a.v, bm.v, acc[t]);
    }
  }
#pragma unroll
  for (int t = 0; t < 4; ++t) {
    const float bn = bias[nb + t * 16 + ln];
    const int d = t * 16 + ln;
#pragma unroll
    for (int i = 0; i < 8; ++i) {
      const int m  = m0 + i + hi * 8;
      const int bb = m >> 10, srow = m & 1023;
      const float vv = acc[t][i] + bn;
      if (z == 2)
        Vt[((size_t)(bb * NH + h) * HS + d) * S_LEN + srow] = (_Float16)vv;
      else {
        _Float16* dst = (z == 0) ? Qh : Kh;
        dst[((size_t)(bb * NH + h) * S_LEN + srow) * HS + d] = (_Float16)vv;
      }
    }
  }
}

// ---------------------------------------------------------------------------
// Stage 2: fused attention.  grid (16 query blocks, 64 bh), 128 threads.
// Per key tile (64):
//   Kd = K_tile(64x64) x E_sub^T(64x128)  -> LDS (cooperative, WMMA)
//   Qd = Q_rows(16x64) x E_sub^T(64x80)   -> LDS (per wave, WMMA)
//   S  = Q K^T (WMMA) + Toeplitz gather of Qd/Kd + mask, *0.125
//   online softmax (shfl_xor row reduce), P via LDS relayout, ctx += P V (WMMA)
// ---------------------------------------------------------------------------
__global__ void __launch_bounds__(128)
attn(const _Float16* __restrict__ Qh, const _Float16* __restrict__ Kh,
     const _Float16* __restrict__ Vt, const _Float16* __restrict__ Eh,
     const float* __restrict__ mask, float* __restrict__ out) {
  __shared__ float    ldsKd[64 * 132];      // padded stride: conflict-free
  __shared__ float    ldsQd[4][16 * 84];
  __shared__ _Float16 ldsP[4][16 * 64];

  const int wave = threadIdx.x >> 5, lane = threadIdx.x & 31;
  const int ln = lane & 15, hi = lane >> 4;
  const int bh = blockIdx.y, b = bh >> 4, h = bh & 15;
  const int l0 = blockIdx.x * 64;
  const int lw = l0 + wave * 16;

  // Q A-operands, K chunks 0..31 / 32..63 (registers, reused by QK and Qd)
  V16H qa[2];
  {
    const _Float16* qrow = Qh + (size_t)(bh * S_LEN + lw + ln) * HS + hi * 16;
    qa[0].q[0] = *(const float4*)(qrow);
    qa[0].q[1] = *(const float4*)(qrow + 8);
    qa[1].q[0] = *(const float4*)(qrow + 32);
    qa[1].q[1] = *(const float4*)(qrow + 40);
  }

  v8f ctx[4] = {};
  float mrow[8], lrow[8];
#pragma unroll
  for (int i = 0; i < 8; ++i) { mrow[i] = -1e30f; lrow[i] = 0.0f; }

  for (int tk = 0; tk < 16; ++tk) {
    const int r0 = tk * 64;
    const int jbase = l0 - r0 + 960;  // E row of j''=0 ; j'' in [0,127]

    __syncthreads();  // previous iteration's ldsKd consumers are done

    // ---- Kd tile: this wave computes rows wave*16..+15 over 128 j'' cols
    {
      V16H ka[2];
      const _Float16* krow =
          Kh + (size_t)(bh * S_LEN + r0 + wave * 16 + ln) * HS + hi * 16;
      ka[0].q[0] = *(const float4*)(krow);
      ka[0].q[1] = *(const float4*)(krow + 8);
      ka[1].q[0] = *(const float4*)(krow + 32);
      ka[1].q[1] = *(const float4*)(krow + 40);
#pragma unroll
      for (int nt = 0; nt < 8; ++nt) {
        const _Float16* erow =
            Eh + (size_t)(jbase + nt * 16 + ln) * HS + hi * 16;
        V16H e0, e1;
        e0.q[0] = *(const float4*)(erow);
        e0.q[1] = *(const float4*)(erow + 8);
        e1.q[0] = *(const float4*)(erow + 32);
        e1.q[1] = *(const float4*)(erow + 40);
        v8f d = {};
        d = wmma_f16(ka[0].v, e0.v, d);
        d = wmma_f16(ka[1].v, e1.v, d);
#pragma unroll
        for (int i = 0; i < 8; ++i)
          ldsKd[(wave * 16 + i + hi * 8) * 132 + nt * 16 + ln] = d[i];
      }
    }

    // ---- Qd per wave: 16 x 80 (uses only c-rr+63 in [0,78])
    {
      const int jbw = jbase + wave * 16;
#pragma unroll
      for (int nt = 0; nt < 5; ++nt) {
        const _Float16* erow = Eh + (size_t)(jbw + nt * 16 + ln) * HS + hi * 16;
        V16H e0, e1;
        e0.q[0] = *(const float4*)(erow);
        e0.q[1] = *(const float4*)(erow + 8);
        e1.q[0] = *(const float4*)(erow + 32);
        e1.q[1] = *(const float4*)(erow + 40);
        v8f d = {};
        d = wmma_f16(qa[0].v, e0.v, d);
        d = wmma_f16(qa[1].v, e1.v, d);
#pragma unroll
        for (int i = 0; i < 8; ++i)
          ldsQd[wave][(i + hi * 8) * 84 + nt * 16 + ln] = d[i];
      }
    }

    // ---- Q K^T : B[k=d, n=r] = K[r,d] (contiguous per lane)
    v8f sc[4];
#pragma unroll
    for (int t = 0; t < 4; ++t) {
      const _Float16* krow =
          Kh + (size_t)(bh * S_LEN + r0 + t * 16 + ln) * HS + hi * 16;
      V16H k0, k1;
      k0.q[0] = *(const float4*)(krow);
      k0.q[1] = *(const float4*)(krow + 8);
      k1.q[0] = *(const float4*)(krow + 32);
      k1.q[1] = *(const float4*)(krow + 40);
      v8f d = {};
      d = wmma_f16(qa[0].v, k0.v, d);
      d = wmma_f16(qa[1].v, k1.v, d);
      sc[t] = d;
    }

    if (tk < 15)  // warm next K tile into caches
      __builtin_prefetch(Kh + (size_t)(bh * S_LEN + r0 + 64 + wave * 16 + ln) * HS, 0, 1);

    __syncthreads();  // ldsKd fully written

    // ---- bias gather + mask + scale, track per-row max
    float smax[8];
#pragma unroll
    for (int i = 0; i < 8; ++i) smax[i] = -1e30f;
#pragma unroll
    for (int t = 0; t < 4; ++t) {
      const int rr = t * 16 + ln;
#pragma unroll
      for (int i = 0; i < 8; ++i) {
        const int c = i + hi * 8;
        const int idx = c - rr + 63;  // Toeplitz shift
        const float bqv = ldsQd[wave][c * 84 + idx];
        const float bkv = ldsKd[rr * 132 + wave * 16 + idx];
        const float mk =
            mask[((size_t)b * S_LEN + (lw + c)) * S_LEN + r0 + rr];
        const float s = (sc[t][i] + bqv + bkv) * 0.125f + mk;
        sc[t][i] = s;
        smax[i] = fmaxf(smax[i], s);
      }
    }
    // row max across the 16-lane half-wave
#pragma unroll
    for (int i = 0; i < 8; ++i) {
      float v = smax[i];
      v = fmaxf(v, __shfl_xor(v, 1));
      v = fmaxf(v, __shfl_xor(v, 2));
      v = fmaxf(v, __shfl_xor(v, 4));
      v = fmaxf(v, __shfl_xor(v, 8));
      smax[i] = v;
    }
    float alpha[8], rsum[8];
#pragma unroll
    for (int i = 0; i < 8; ++i) {
      const float mnew = fmaxf(mrow[i], smax[i]);
      alpha[i] = __expf(mrow[i] - mnew);
      mrow[i] = mnew;
      rsum[i] = 0.0f;
    }
    // P = exp(S - m), stage to LDS in [c][rr] order
#pragma unroll
    for (int t = 0; t < 4; ++t) {
#pragma unroll
      for (int i = 0; i < 8; ++i) {
        const int c = i + hi * 8;
        const float p = __expf(sc[t][i] - mrow[i]);
        rsum[i] += p;
        ldsP[wave][c * 64 + t * 16 + ln] = (_Float16)p;
      }
    }
#pragma unroll
    for (int i = 0; i < 8; ++i) {
      float v = rsum[i];
      v += __shfl_xor(v, 1);
      v += __shfl_xor(v, 2);
      v += __shfl_xor(v, 4);
      v += __shfl_xor(v, 8);
      lrow[i] = lrow[i] * alpha[i] + v;
#pragma unroll
      for (int t = 0; t < 4; ++t) ctx[t][i] *= alpha[i];
    }

    // ---- ctx += P V : P re-read from LDS in A layout; B from transposed V
    V16H pa[2];
    {
      const _Float16* prow = &ldsP[wave][ln * 64 + hi * 16];
      pa[0].q[0] = *(const float4*)(prow);
      pa[0].q[1] = *(const float4*)(prow + 8);
      pa[1].q[0] = *(const float4*)(prow + 32);
      pa[1].q[1] = *(const float4*)(prow + 40);
    }
#pragma unroll
    for (int t = 0; t < 4; ++t) {
      const _Float16* vrow =
          Vt + ((size_t)bh * HS + t * 16 + ln) * S_LEN + r0 + hi * 16;
      V16H v0, v1;
      v0.q[0] = *(const float4*)(vrow);
      v0.q[1] = *(const float4*)(vrow + 8);
      v1.q[0] = *(const float4*)(vrow + 32);
      v1.q[1] = *(const float4*)(vrow + 40);
      ctx[t] = wmma_f16(pa[0].v, v0.v, ctx[t]);
      ctx[t] = wmma_f16(pa[1].v, v1.v, ctx[t]);
    }
  }

  // ---- epilogue: normalize and store f32 output [B,S,nh*hs]
#pragma unroll
  for (int i = 0; i < 8; ++i) {
    const float inv = 1.0f / lrow[i];
    const int srow = lw + i + hi * 8;
#pragma unroll
    for (int t = 0; t < 4; ++t)
      out[((size_t)b * S_LEN + srow) * HDIM + h * HS + t * 16 + ln] =
          ctx[t][i] * inv;
  }
}

// ---------------------------------------------------------------------------
extern "C" void kernel_launch(void* const* d_in, const int* in_sizes, int n_in,
                              void* d_out, int out_size, void* d_ws,
                              size_t ws_size, hipStream_t stream) {
  (void)in_sizes; (void)n_in; (void)out_size; (void)ws_size;
  const float* hidden = (const float*)d_in[0];
  const float* amask  = (const float*)d_in[1];
  const float* Wq     = (const float*)d_in[2];
  const float* bq     = (const float*)d_in[3];
  const float* Wk     = (const float*)d_in[4];
  const float* bk     = (const float*)d_in[5];
  const float* Wv     = (const float*)d_in[6];
  const float* bv     = (const float*)d_in[7];
  const float* dist   = (const float*)d_in[8];
  float* out = (float*)d_out;
  _Float16* ws = (_Float16*)d_ws;

  auto cvt = [&](const float* s, _Float16* d, int n, int npad) {
    cvt_f32_f16<<<(npad + 255) / 256, 256, 0, stream>>>(s, d, n, npad);
  };
  cvt(hidden, ws + XH_OFF, 4096 * 1024, 4096 * 1024);
  cvt(Wq,     ws + WQ_OFF, 1024 * 1024, 1024 * 1024);
  cvt(Wk,     ws + WK_OFF, 1024 * 1024, 1024 * 1024);
  cvt(Wv,     ws + WV_OFF, 1024 * 1024, 1024 * 1024);
  cvt(dist,   ws + E_OFF,  2047 * 64,   2048 * 64);  // +1 zero pad row

  qkv_gemm<<<dim3(64, 16, 3), 128, 0, stream>>>(
      ws + XH_OFF, ws + WQ_OFF, ws + WK_OFF, ws + WV_OFF, bq, bk, bv,
      ws + QH_OFF, ws + KH_OFF, ws + VT_OFF);

  attn<<<dim3(16, 64), 128, 0, stream>>>(ws + QH_OFF, ws + KH_OFF,
                                         ws + VT_OFF, ws + E_OFF, amask, out);
}
